// MassivePool_38697655336981
// MI455X (gfx1250) — matrix operand order repaired and checked by reading, not compiled, not run on hardware
//
#include <hip/hip_runtime.h>
#include <hip/hip_fp16.h>
#include <math.h>

typedef _Float16 v16h __attribute__((ext_vector_type(16)));
typedef _Float16 v8h  __attribute__((ext_vector_type(8)));
typedef float    v8f  __attribute__((ext_vector_type(8)));

#define NPOOL   200000
#define DIM     512
#define NQ      1024      // B*S
#define TOPK    32
#define NSLICES 50
#define SLICE   4000      // NPOOL / NSLICES
#define CHUNK   64
#define MPANEL  128
#define NPANELS 8         // NQ / MPANEL
#define QPAD    520       // padded LDS row stride in halves (bank-conflict break)
#define SPAD    65        // padded LDS row stride in floats
#define NCAND   (NSLICES * TOPK)   // 1600

// LDS offset of a __shared__ object: generic (flat) pointers into the LDS
// aperture carry the AS3 offset in the low 32 bits (flat addr = {aperture_hi,
// lds_offset}), which is exactly what async-load VDST expects.
__device__ __forceinline__ unsigned lds_off32(const void* p) {
    return (unsigned)(unsigned long long)p;
}

// CDNA5 async global->LDS copy, 16B per lane, tracked with ASYNCcnt.
__device__ __forceinline__ void async_load_lds_b128(unsigned lds_byte_off,
                                                    const void* gaddr) {
    unsigned long long ga = (unsigned long long)gaddr;
    asm volatile("global_load_async_to_lds_b128 %0, %1, off"
                 :: "v"(lds_byte_off), "v"(ga) : "memory");
}

__device__ __forceinline__ void wait_asynccnt0() {
    asm volatile("s_wait_asynccnt 0x0" ::: "memory");
}

// ---------------------------------------------------------------------------
// Kernel 1: fused scores GEMM (f16 WMMA, f32 acc) + per-slice partial top-32.
// grid = (NSLICES, NPANELS), block = 256 (8 wave32).
// 128 queries resident in LDS (f16) so the 410 MB key matrix streams from HBM
// only NPANELS=8 times (~3.3 GB ~ 140 us at 23.3 TB/s); WMMA f16 keeps the
// 210 GFLOP score GEMM off the critical path.
// ---------------------------------------------------------------------------
__global__ __launch_bounds__(256)
void mp_scores_topk(const float* __restrict__ query,
                    const float* __restrict__ keys,
                    float* __restrict__ ps, int* __restrict__ pidx)
{
    __shared__ _Float16 qbuf[MPANEL * QPAD];   // 133 KB
    __shared__ _Float16 kbuf[CHUNK  * QPAD];   //  67 KB
    __shared__ float    sbuf[MPANEL * SPAD];   //  33 KB
    __shared__ float    ts[MPANEL * TOPK];     //  16 KB
    __shared__ int      ti[MPANEL * TOPK];     //  16 KB

    const int slice   = blockIdx.x;
    const int panel   = blockIdx.y;
    const int q0      = panel * MPANEL;
    const int n_start = slice * SLICE;
    const int n_end   = n_start + SLICE;
    const int tid     = threadIdx.x;
    const int wave    = tid >> 5;
    const int lane    = tid & 31;

    // Load 128-query panel, convert f32 -> f16 into LDS.
    {
        const float4* qp = (const float4*)(query + (size_t)q0 * DIM);
        for (int i = tid; i < MPANEL * DIM / 4; i += 256) {
            int row = i >> 7;          // DIM/4 == 128
            int c4  = i & 127;
            float4 v = qp[i];
            _Float16* dst = qbuf + row * QPAD + c4 * 4;
            dst[0] = (_Float16)v.x; dst[1] = (_Float16)v.y;
            dst[2] = (_Float16)v.z; dst[3] = (_Float16)v.w;
        }
    }
    for (int i = tid; i < MPANEL * TOPK; i += 256) { ts[i] = -3.0e38f; ti[i] = -1; }
    __syncthreads();

    const int nchunks = (SLICE + CHUNK - 1) / CHUNK;
    for (int c = 0; c < nchunks; ++c) {
        const int nbase = n_start + c * CHUNK;

        // Load 64-key chunk, f32 -> f16 into LDS (zeros past slice end).
        for (int i = tid; i < CHUNK * DIM / 4; i += 256) {
            int row = i >> 7;
            int c4  = i & 127;
            int n   = nbase + row;
            float4 v = make_float4(0.f, 0.f, 0.f, 0.f);
            if (n < n_end) v = ((const float4*)(keys + (size_t)n * DIM))[c4];
            _Float16* dst = kbuf + row * QPAD + c4 * 4;
            dst[0] = (_Float16)v.x; dst[1] = (_Float16)v.y;
            dst[2] = (_Float16)v.z; dst[3] = (_Float16)v.w;
        }
        __syncthreads();

        // WMMA: wave w owns M-tile w (16 queries) x 4 N-tiles (64 keys).
        {
            const int arow = wave * 16 + (lane & 15);
            const int hiA  = (lane >> 4) << 4;   // 0 or 16 (A: K 0..15 / 16..31)
            const int hiB  = (lane >> 4) << 4;   // 0 or 16 (B: K 0..15 / 16..31)
            v8f acc[4] = {};
            for (int kk = 0; kk < DIM; kk += 32) {
                const _Float16* qr = qbuf + arow * QPAD + kk + hiA;
                v8h alo = *(const v8h*)qr;
                v8h ahi = *(const v8h*)(qr + 8);
                v16h a;
#pragma unroll
                for (int e = 0; e < 8; ++e) { a[e] = alo[e]; a[e + 8] = ahi[e]; }
#pragma unroll
                for (int j = 0; j < 4; ++j) {
                    const _Float16* kr = kbuf + (j * 16 + (lane & 15)) * QPAD + kk + hiB;
                    v16h b = *(const v16h*)kr;
                    acc[j] = __builtin_amdgcn_wmma_f32_16x16x32_f16(
                        false, a, false, b, (short)0, acc[j], false, false);
                }
            }
            // C layout: acc[r] -> M = r + 8*(lane>=16), N = lane&15
            const int mbase = wave * 16 + ((lane >> 4) << 3);
            const int ncol  = lane & 15;
#pragma unroll
            for (int r = 0; r < 8; ++r) {
#pragma unroll
                for (int j = 0; j < 4; ++j)
                    sbuf[(mbase + r) * SPAD + j * 16 + ncol] = acc[j][r];
            }
        }
        __syncthreads();

        // Top-32 insertion update: one thread per query (sorted descending).
        if (tid < MPANEL) {
            float*       tsq  = ts + tid * TOPK;
            int*         tiq  = ti + tid * TOPK;
            const float* srow = sbuf + tid * SPAD;
            int jmax = CHUNK;
            if (nbase + CHUNK > n_end) jmax = n_end - nbase;
            for (int j = 0; j < jmax; ++j) {
                float s = srow[j];
                if (s > tsq[TOPK - 1]) {
                    int p = TOPK - 1;
                    while (p > 0 && tsq[p - 1] < s) {
                        tsq[p] = tsq[p - 1]; tiq[p] = tiq[p - 1]; --p;
                    }
                    tsq[p] = s; tiq[p] = nbase + j;
                }
            }
        }
        __syncthreads();
    }

    // Emit partial top-32 for this (query, slice).
    for (int i = tid; i < MPANEL * TOPK; i += 256) {
        int qi = i >> 5;
        int k  = i & 31;
        size_t off = (((size_t)(q0 + qi)) * NSLICES + slice) * TOPK + k;
        ps[off]   = ts[i];
        pidx[off] = ti[i];
    }
}

// ---------------------------------------------------------------------------
// Kernel 2: merge 50 partial lists -> global top-32, softmax, gather pool,
// weighted aggregate. grid = NQ blocks, 256 threads.
// Candidate lists are pulled into LDS with CDNA5 async-to-LDS copies
// (ASYNCcnt) -- raw f32/i32, no conversion needed, bypasses VGPR staging.
// ---------------------------------------------------------------------------
__global__ __launch_bounds__(256)
void mp_merge_gather(const float* __restrict__ ps, const int* __restrict__ pidx,
                     const float* __restrict__ pool, float* __restrict__ agg)
{
    __shared__ float cs[NCAND];
    __shared__ int   ci[NCAND];
    __shared__ float red_s[256];
    __shared__ int   red_i[256];
    __shared__ float sel_s[TOPK];
    __shared__ int   sel_i[TOPK];
    __shared__ float w[TOPK];

    const int q   = blockIdx.x;
    const int tid = threadIdx.x;

    // Async global->LDS: 400 x b128 for scores, 400 x b128 for indices.
    for (int i = tid * 4; i < NCAND; i += 256 * 4) {
        async_load_lds_b128(lds_off32(&cs[i]), ps   + (size_t)q * NCAND + i);
        async_load_lds_b128(lds_off32(&ci[i]), pidx + (size_t)q * NCAND + i);
    }
    wait_asynccnt0();
    __syncthreads();

    // 32 rounds of parallel argmax over 1600 candidates.
    for (int k = 0; k < TOPK; ++k) {
        float best = -3.0e38f; int bi = -1;
        for (int i = tid; i < NCAND; i += 256) {
            float v = cs[i];
            if (v > best) { best = v; bi = i; }
        }
        red_s[tid] = best; red_i[tid] = bi;
        __syncthreads();
        for (int off = 128; off > 0; off >>= 1) {
            if (tid < off && red_s[tid + off] > red_s[tid]) {
                red_s[tid] = red_s[tid + off]; red_i[tid] = red_i[tid + off];
            }
            __syncthreads();
        }
        if (tid == 0) {
            int p = red_i[0];
            sel_s[k] = red_s[0];
            sel_i[k] = (p >= 0) ? ci[p] : 0;
            if (p >= 0) cs[p] = -3.4e38f;
        }
        __syncthreads();
    }

    if (tid == 0) {
        float mx = sel_s[0];            // sorted descending
        float sum = 0.f;
        for (int k = 0; k < TOPK; ++k) { float e = __expf(sel_s[k] - mx); w[k] = e; sum += e; }
        float inv = 1.0f / sum;
        for (int k = 0; k < TOPK; ++k) w[k] *= inv;
    }
    __syncthreads();

    for (int d = tid; d < DIM; d += 256) {
        float a = 0.f;
#pragma unroll 4
        for (int k = 0; k < TOPK; ++k) {
            int idx = sel_i[k]; if (idx < 0) idx = 0;
            a += w[k] * pool[(size_t)idx * DIM + d];
        }
        agg[(size_t)q * DIM + d] = a;
    }
}

// ---------------------------------------------------------------------------
// Kernel 3: out = agg @ W_out^T via f16 WMMA.
// grid = (NQ/16, 4), block = 256: wave w of block (mt, gy) owns tile
// (M = mt*16.., O = (gy*8+w)*16..).
// ---------------------------------------------------------------------------
__global__ __launch_bounds__(256)
void mp_out_gemm(const float* __restrict__ agg, const float* __restrict__ Wout,
                 float* __restrict__ out)
{
    __shared__ _Float16 abuf[16 * QPAD];

    const int mt   = blockIdx.x;
    const int tid  = threadIdx.x;
    const int wave = tid >> 5;
    const int lane = tid & 31;

    const float4* ap = (const float4*)(agg + (size_t)mt * 16 * DIM);
    for (int i = tid; i < 16 * DIM / 4; i += 256) {
        int row = i >> 7;
        int c4  = i & 127;
        float4 v = ap[i];
        _Float16* dst = abuf + row * QPAD + c4 * 4;
        dst[0] = (_Float16)v.x; dst[1] = (_Float16)v.y;
        dst[2] = (_Float16)v.z; dst[3] = (_Float16)v.w;
    }
    __syncthreads();

    const int nt   = blockIdx.y * 8 + wave;        // 0..31
    const int ocol = nt * 16 + (lane & 15);        // W_out row == output column
    const int hiA  = (lane >> 4) << 4;
    const int hiB  = (lane >> 4) << 4;
    const int arow = lane & 15;

    v8f acc = {};
    for (int kk = 0; kk < DIM; kk += 32) {
        const _Float16* ar = abuf + arow * QPAD + kk + hiA;
        v8h alo = *(const v8h*)ar;
        v8h ahi = *(const v8h*)(ar + 8);
        v16h a;
#pragma unroll
        for (int e = 0; e < 8; ++e) { a[e] = alo[e]; a[e + 8] = ahi[e]; }

        const float* wr = Wout + (size_t)ocol * DIM + kk + hiB;
        float4 w0 = *(const float4*)(wr);
        float4 w1 = *(const float4*)(wr + 4);
        float4 w2 = *(const float4*)(wr + 8);
        float4 w3 = *(const float4*)(wr + 12);
        v16h b;
        b[0]  = (_Float16)w0.x; b[1]  = (_Float16)w0.y; b[2]  = (_Float16)w0.z; b[3]  = (_Float16)w0.w;
        b[4]  = (_Float16)w1.x; b[5]  = (_Float16)w1.y; b[6]  = (_Float16)w1.z; b[7]  = (_Float16)w1.w;
        b[8]  = (_Float16)w2.x; b[9]  = (_Float16)w2.y; b[10] = (_Float16)w2.z; b[11] = (_Float16)w2.w;
        b[12] = (_Float16)w3.x; b[13] = (_Float16)w3.y; b[14] = (_Float16)w3.z; b[15] = (_Float16)w3.w;

        acc = __builtin_amdgcn_wmma_f32_16x16x32_f16(
            false, a, false, b, (short)0, acc, false, false);
    }

    const int mrow = mt * 16 + ((lane >> 4) << 3);
    const int obase = nt * 16 + (lane & 15);
#pragma unroll
    for (int r = 0; r < 8; ++r)
        out[(size_t)(mrow + r) * DIM + obase] = acc[r];
}

// ---------------------------------------------------------------------------
extern "C" void kernel_launch(void* const* d_in, const int* in_sizes, int n_in,
                              void* d_out, int out_size, void* d_ws, size_t ws_size,
                              hipStream_t stream)
{
    const float* query = (const float*)d_in[0];   // [2,512,512]
    const float* keys  = (const float*)d_in[1];   // [200000,512]
    const float* pool  = (const float*)d_in[2];   // [200000,512]
    const float* Wout  = (const float*)d_in[3];   // [512,512]
    float* out = (float*)d_out;                   // [2,512,512]

    char*  ws   = (char*)d_ws;
    float* ps   = (float*)ws;                                           // NQ*NCAND f32 (6.6 MB)
    int*   pidx = (int*)(ws + sizeof(float) * (size_t)NQ * NCAND);      // NQ*NCAND i32 (6.6 MB)
    float* agg  = (float*)(ws + 2 * sizeof(float) * (size_t)NQ * NCAND);// NQ*DIM f32 (2 MB)

    mp_scores_topk <<<dim3(NSLICES, NPANELS), 256, 0, stream>>>(query, keys, ps, pidx);
    mp_merge_gather<<<dim3(NQ),               256, 0, stream>>>(ps, pidx, pool, agg);
    mp_out_gemm    <<<dim3(NQ / 16, 4),       256, 0, stream>>>(agg, Wout, out);
}
